// ResidualAttentionBlock_16243566313905
// MI455X (gfx1250) — matrix operand order repaired and checked by reading, not compiled, not run on hardware
//
#include <hip/hip_runtime.h>
#include <math.h>

#define BB_ 4
#define TT_ 1000
#define TP_ 1008          // tokens padded to multiple of 16
#define DD_ 256
#define HH_ 8
#define SSTR 268          // LDS row stride in halves (134 dwords -> conflict-free)

typedef __attribute__((ext_vector_type(16))) _Float16 v16h;
typedef __attribute__((ext_vector_type(8)))  float    v8f;
typedef __attribute__((ext_vector_type(2)))  _Float16 h2;

union U16 { v16h v; unsigned u[8]; };
union H4  { uint2 v; h2 h[2]; };

static __device__ __forceinline__ unsigned pack2f(float a, float b) {
  union { _Float16 h[2]; unsigned u; } t;
  t.h[0] = (_Float16)a; t.h[1] = (_Float16)b; return t.u;
}
static __device__ __forceinline__ void unpack2f(unsigned u, float& a, float& b) {
  union { unsigned u; _Float16 h[2]; } t; t.u = u;
  a = (float)t.h[0]; b = (float)t.h[1];
}
static __device__ __forceinline__ h2 habs2(h2 a) {
  union { h2 h; unsigned u; } t; t.h = a; t.u &= 0x7FFF7FFFu; return t.h;
}
// A-matrix (16x32 f16) documented interleaved k-index for register j, lane-half hf
static __device__ __forceinline__ int kidxA(int j, int hf) {
  return ((j >> 2) << 4) + ((j & 3) << 1) + (hf << 3);
}

// ---------------------------------------------------------------------------
// K1: QV projection.  out[m,n] = sum_k x[m,k] * wqv[n,k] + bias[n]
// q  -> f16 [B][H][TP][D]   (only feeds the L1 kernel)
// vf -> f16 transposed [B][H][D][TP]
// vb -> f16 transposed [B][H][D][TP]
// ---------------------------------------------------------------------------
__global__ void k_proj(const float* __restrict__ x, const float* __restrict__ wqv,
                       const float* __restrict__ bias,
                       _Float16* __restrict__ q, _Float16* __restrict__ vfT,
                       _Float16* __restrict__ vbT) {
  const int MT = 250, NT = 384;
  int wave = blockIdx.x * (blockDim.x >> 5) + (threadIdx.x >> 5);
  if (wave >= MT * NT) return;
  int mt = wave / NT, nt = wave % NT;
  int m0 = mt * 16, n0 = nt * 16;
  int lane = threadIdx.x & 31;
  int col = lane & 15, hf = lane >> 4;

  const float* xrow = x + (size_t)(m0 + col) * DD_;              // A row
  const int n = n0 + col;                                        // B column
  const float* wcol = wqv + (size_t)n * DD_;

  v8f acc = {};
  U16 a, b;
#pragma unroll
  for (int ks = 0; ks < 8; ++ks) {
    int k0 = ks * 32;
#pragma unroll
    for (int j = 0; j < 8; ++j) {
      int ka = k0 + kidxA(j, hf);
      a.u[j] = pack2f(xrow[ka], xrow[ka + 1]);
      int kb = k0 + kidxA(j, hf);
      b.u[j] = pack2f(wcol[kb], wcol[kb + 1]);
    }
    acc = __builtin_amdgcn_wmma_f32_16x16x32_f16(false, a.v, false, b.v,
                                                 (short)0, acc, false, false);
  }

  int c = n >> 8;          // 0..23 : q heads, vf heads, vb heads
  int w = n & 255;
  float bn = bias[n];
#pragma unroll
  for (int r = 0; r < 8; ++r) {
    int mg = m0 + r + (hf << 3);          // flat token row
    int bb = mg / TT_, tk = mg - bb * TT_;
    float val = acc[r] + bn;
    if (c < 8) {
      q[(((size_t)bb * HH_ + c) * TP_ + tk) * DD_ + w] = (_Float16)val;
    } else if (c < 16) {
      vfT[(((size_t)bb * HH_ + (c - 8)) * DD_ + w) * TP_ + tk] = (_Float16)val;
    } else {
      vbT[(((size_t)bb * HH_ + (c - 16)) * DD_ + w) * TP_ + tk] = (_Float16)val;
    }
  }
}

// ---------------------------------------------------------------------------
// K2: L1-distance scores + exp (dominant kernel, VALU bound).
// S[s,d] = exp(-sum_w |q[s,w]-k[d,w]| / 16),  k[d,w] = x[b,d,w]*wk[h,w].
// Packed-f16 inner loop (v_pk_add_f16): 3 VALU per 2 elements; f16 partial
// sums flushed to f32 every 64 w.  Tile 64s x 64d, LDS f16 (68.6 KB dyn).
// ---------------------------------------------------------------------------
__global__ void k_scores(const _Float16* __restrict__ q, const float* __restrict__ x,
                         const float* __restrict__ wk,
                         _Float16* __restrict__ S, _Float16* __restrict__ ST) {
  extern __shared__ _Float16 smem_h[];
  _Float16* qs  = smem_h;                 // [64][SSTR]
  _Float16* ksm = smem_h + 64 * SSTR;     // [64][SSTR]

  const int bh = blockIdx.z;
  const int bb = bh / HH_, hh = bh % HH_;
  const int s0 = blockIdx.y * 64;
  const int d0 = blockIdx.x * 64;

  for (int i = threadIdx.x; i < 64 * 256; i += 256) {
    int row = i >> 8, cl = i & 255;
    int sg = s0 + row;
    qs[row * SSTR + cl] =
        (sg < TP_) ? q[((size_t)bh * TP_ + sg) * DD_ + cl] : (_Float16)0.f;
    int dg = d0 + row;
    float kv =
        (dg < TT_) ? x[((size_t)bb * TT_ + dg) * DD_ + cl] * wk[hh * DD_ + cl] : 0.f;
    ksm[row * SSTR + cl] = (_Float16)kv;
  }
  __syncthreads();

  const int si = (threadIdx.x & 15) * 4;
  const int dj = (threadIdx.x >> 4) * 4;
  float accf[4][4] = {};
  for (int wo = 0; wo < 256; wo += 64) {          // f32 flush chunk
    h2 acch[4][4] = {};
    for (int w = wo; w < wo + 64; w += 4) {       // ds_load_b64 step
      H4 qv[4], kv[4];
#pragma unroll
      for (int i = 0; i < 4; ++i) qv[i].v = *(const uint2*)(qs + (si + i) * SSTR + w);
#pragma unroll
      for (int j = 0; j < 4; ++j) kv[j].v = *(const uint2*)(ksm + (dj + j) * SSTR + w);
#pragma unroll
      for (int i = 0; i < 4; ++i)
#pragma unroll
        for (int j = 0; j < 4; ++j) {
          acch[i][j] += habs2(qv[i].h[0] - kv[j].h[0]);
          acch[i][j] += habs2(qv[i].h[1] - kv[j].h[1]);
        }
    }
#pragma unroll
    for (int i = 0; i < 4; ++i)
#pragma unroll
      for (int j = 0; j < 4; ++j)
        accf[i][j] += (float)acch[i][j][0] + (float)acch[i][j][1];
  }

#pragma unroll
  for (int i = 0; i < 4; ++i) {
    int sg = s0 + si + i;
#pragma unroll
    for (int j = 0; j < 4; ++j) {
      int dg = d0 + dj + j;
      if (sg < TP_ && dg < TP_) {
        float sv = __expf(accf[i][j] * -0.0625f);
        _Float16 hv = (sg < TT_ && dg < TT_) ? (_Float16)sv : (_Float16)0.f;
        S [((size_t)bh * TP_ + sg) * TP_ + dg] = hv;
        ST[((size_t)bh * TP_ + dg) * TP_ + sg] = hv;
      }
    }
  }
}

// ---------------------------------------------------------------------------
// K3: softmax denominators.  Z[bh,d] = 1 + sum_s S[s,d]  (row sum over ST)
// ---------------------------------------------------------------------------
__global__ void k_z(const _Float16* __restrict__ ST, float* __restrict__ invZ) {
  int idx = blockIdx.x * blockDim.x + threadIdx.x;
  if (idx >= BB_ * HH_ * TP_) return;
  int bh = idx / TP_, dd = idx % TP_;
  const unsigned* row = (const unsigned*)(ST + ((size_t)bh * TP_ + dd) * TP_);
  float sum = 1.0f;
  for (int i = 0; i < TP_ / 2; ++i) {
    float a, b; unpack2f(row[i], a, b);
    sum += a + b;
  }
  invZ[idx] = 1.0f / sum;
}

// ---------------------------------------------------------------------------
// K4: fold invZ[s] into vbT:  vbT[bh][w][s] *= invZ[bh][s]
// ---------------------------------------------------------------------------
__global__ void k_scalev(_Float16* __restrict__ vbT, const float* __restrict__ invZ) {
  int idx = blockIdx.x * blockDim.x + threadIdx.x;
  const int NHALF2 = BB_ * HH_ * DD_ * (TP_ / 2);
  if (idx >= NHALF2) return;
  int s2 = idx % (TP_ / 2);
  int t1 = idx / (TP_ / 2);
  int w = t1 % DD_, bh = t1 / DD_;
  int s = s2 * 2;
  unsigned* p = (unsigned*)(vbT + (((size_t)bh * DD_ + w) * TP_ + s));
  float a, b; unpack2f(*p, a, b);
  a *= invZ[bh * TP_ + s];
  b *= invZ[bh * TP_ + s + 1];
  *p = pack2f(a, b);
}

// ---------------------------------------------------------------------------
// K5: attention apply.  O[b,d,w] = sum_h ( invZ_h[d] * (ST@vf)[d,w] + (S@vb')[d,w] )
// One wave per (b, d-tile, w-tile); loops 8 heads, K = 1008 (32 WMMA steps x2).
// ---------------------------------------------------------------------------
__global__ void k_attn(const _Float16* __restrict__ S, const _Float16* __restrict__ ST,
                       const _Float16* __restrict__ vfT, const _Float16* __restrict__ vbT,
                       const float* __restrict__ invZ, float* __restrict__ O) {
  const int DTL = TP_ / 16, WTL = DD_ / 16;    // 63, 16
  int wave = blockIdx.x * (blockDim.x >> 5) + (threadIdx.x >> 5);
  if (wave >= BB_ * DTL * WTL) return;
  int bb = wave / (DTL * WTL);
  int rem = wave % (DTL * WTL);
  int dt = rem / WTL, wt = rem % WTL;
  int m0 = dt * 16, n0 = wt * 16;
  int lane = threadIdx.x & 31;
  int col = lane & 15, hf = lane >> 4;

  float accO[8] = {0.f, 0.f, 0.f, 0.f, 0.f, 0.f, 0.f, 0.f};

  for (int h = 0; h < HH_; ++h) {
    int bh = bb * HH_ + h;
    const _Float16* a1row = ST + ((size_t)bh * TP_ + m0 + col) * TP_;
    const _Float16* a2row = S  + ((size_t)bh * TP_ + m0 + col) * TP_;
    const _Float16* b1col = vfT + (((size_t)bh * DD_ + n0 + col) * TP_) + (hf << 4);
    const _Float16* b2col = vbT + (((size_t)bh * DD_ + n0 + col) * TP_) + (hf << 4);

    v8f acc1 = {}, acc2 = {};
    U16 a1, a2, b1, b2;
    for (int kt = 0; kt < TP_ / 16; kt += 2) {
      int k0 = kt * 16;
#pragma unroll
      for (int j = 0; j < 8; ++j) {
        int ka = k0 + kidxA(j, hf);
        a1.u[j] = *(const unsigned*)(a1row + ka);
        a2.u[j] = *(const unsigned*)(a2row + ka);
        int kb = k0 + kidxA(j, hf) - (hf << 4);
        b1.u[j] = *(const unsigned*)(b1col + kb);
        b2.u[j] = *(const unsigned*)(b2col + kb);
      }
      acc1 = __builtin_amdgcn_wmma_f32_16x16x32_f16(false, a1.v, false, b1.v,
                                                    (short)0, acc1, false, false);
      acc2 = __builtin_amdgcn_wmma_f32_16x16x32_f16(false, a2.v, false, b2.v,
                                                    (short)0, acc2, false, false);
    }
    const float* iz = invZ + bh * TP_;
#pragma unroll
    for (int r = 0; r < 8; ++r)
      accO[r] += iz[m0 + r + (hf << 3)] * acc1[r] + acc2[r];
  }

#pragma unroll
  for (int r = 0; r < 8; ++r) {
    int dg = m0 + r + (hf << 3);
    if (dg < TT_) O[((size_t)bb * TT_ + dg) * DD_ + n0 + col] = accO[r];
  }
}

// ---------------------------------------------------------------------------
// K6: y = O*sigmoid(1.702*O);  out = x + y @ fanin_w.T + fanin_b
// ---------------------------------------------------------------------------
__global__ void k_final(const float* __restrict__ O, const float* __restrict__ fw,
                        const float* __restrict__ fb, const float* __restrict__ x,
                        float* __restrict__ out) {
  const int MT = 250, NT = 16;
  int wave = blockIdx.x * (blockDim.x >> 5) + (threadIdx.x >> 5);
  if (wave >= MT * NT) return;
  int mt = wave / NT, nt = wave % NT;
  int m0 = mt * 16, n0 = nt * 16;
  int lane = threadIdx.x & 31;
  int col = lane & 15, hf = lane >> 4;

  const float* orow = O + (size_t)(m0 + col) * DD_;
  const float* wcol = fw + (size_t)(n0 + col) * DD_;

  v8f acc = {};
  U16 a, b;
#pragma unroll
  for (int ks = 0; ks < 8; ++ks) {
    int k0 = ks * 32;
#pragma unroll
    for (int j = 0; j < 8; ++j) {
      int ka = k0 + kidxA(j, hf);
      float o0 = orow[ka], o1 = orow[ka + 1];
      float s0 = o0 / (1.0f + __expf(-1.702f * o0));
      float s1 = o1 / (1.0f + __expf(-1.702f * o1));
      a.u[j] = pack2f(s0, s1);
      int kb = k0 + kidxA(j, hf);
      b.u[j] = pack2f(wcol[kb], wcol[kb + 1]);
    }
    acc = __builtin_amdgcn_wmma_f32_16x16x32_f16(false, a.v, false, b.v,
                                                 (short)0, acc, false, false);
  }

  int n = n0 + col;
  float bn = fb[n];
#pragma unroll
  for (int r = 0; r < 8; ++r) {
    int mg = m0 + r + (hf << 3);
    out[(size_t)mg * DD_ + n] = x[(size_t)mg * DD_ + n] + bn + acc[r];
  }
}

// ---------------------------------------------------------------------------
extern "C" void kernel_launch(void* const* d_in, const int* in_sizes, int n_in,
                              void* d_out, int out_size, void* d_ws, size_t ws_size,
                              hipStream_t stream) {
  const float* x    = (const float*)d_in[0];
  const float* wqvw = (const float*)d_in[1];
  const float* wqvb = (const float*)d_in[2];
  const float* wk   = (const float*)d_in[3];
  const float* finw = (const float*)d_in[4];
  const float* finb = (const float*)d_in[5];
  float* out = (float*)d_out;

  char* ws = (char*)d_ws;
  const size_t QB  = (size_t)BB_ * HH_ * TP_ * DD_ * 2;     // q   f16
  const size_t VTB = (size_t)BB_ * HH_ * DD_ * TP_ * 2;     // vfT/vbT f16
  const size_t SB  = (size_t)BB_ * HH_ * TP_ * TP_ * 2;     // S/ST f16
  const size_t ZB  = (size_t)BB_ * HH_ * TP_ * 4;           // invZ f32

  _Float16* q    = (_Float16*)ws;                 ws += QB;
  _Float16* vfT  = (_Float16*)ws;                 ws += VTB;
  _Float16* vbT  = (_Float16*)ws;                 ws += VTB;
  _Float16* S    = (_Float16*)ws;                 ws += SB;
  _Float16* ST   = (_Float16*)ws;                 ws += SB;
  float*    invZ = (float*)ws;                    ws += ZB;
  float*    O    = (float*)ws;

  // 1. projection GEMM (WMMA)
  k_proj<<<12000, 256, 0, stream>>>(x, wqvw, wqvb, q, vfT, vbT);

  // 2. L1 scores + exp (packed-f16 VALU; dominant kernel)
  const int SMEM = 2 * 64 * SSTR * 2;  // 68608 B dynamic LDS -> 4 blocks/WGP
  (void)hipFuncSetAttribute((const void*)k_scores,
                            hipFuncAttributeMaxDynamicSharedMemorySize, SMEM);
  k_scores<<<dim3(16, 16, BB_ * HH_), 256, SMEM, stream>>>(q, x, wk, S, ST);

  // 3. denominators (sink logit +1) and reciprocal
  k_z<<<(BB_ * HH_ * TP_ + 255) / 256, 256, 0, stream>>>(ST, invZ);

  // 4. fold invZ[s] into vb
  {
    int n2 = BB_ * HH_ * DD_ * (TP_ / 2);
    k_scalev<<<(n2 + 255) / 256, 256, 0, stream>>>(vbT, invZ);
  }

  // 5. attention apply (WMMA, both einsums fused)
  k_attn<<<(BB_ * (TP_ / 16) * (DD_ / 16) + 7) / 8, 256, 0, stream>>>(
      S, ST, vfT, vbT, invZ, O);

  // 6. QuickGELU + fan-in GEMM + residual (WMMA)
  k_final<<<(250 * 16 + 7) / 8, 256, 0, stream>>>(O, finw, finb, x, out);
}
